// TapNet_34179349741867
// MI455X (gfx1250) — matrix operand order repaired and checked by reading, not compiled, not run on hardware
//
#include <hip/hip_runtime.h>
#include <hip/hip_bf16.h>

typedef __bf16 bf16;
typedef bf16 v16bf __attribute__((ext_vector_type(16)));
typedef bf16 v8bf  __attribute__((ext_vector_type(8)));
typedef bf16 v4bf  __attribute__((ext_vector_type(4)));
typedef float v8f  __attribute__((ext_vector_type(8)));
typedef unsigned v4u __attribute__((ext_vector_type(4)));
typedef int v8i __attribute__((ext_vector_type(8)));
typedef int v4i __attribute__((ext_vector_type(4)));

#define DEVI __device__ __forceinline__

constexpr int N = 262144, NFEAT = 256, H = 512, DOUT = 128, C = 32, D = 128;
constexpr int NTR = 131072;
constexpr float EPS = 1e-5f, SLOPE = 0.01f;

constexpr int MT  = 32;           // rows per workgroup tile
constexpr int LDA = NFEAT + 8;    // LDS leading dim for x tile (bf16), 528B rows
constexpr int LDH = H + 8;        // LDS leading dim for h tile (bf16)
constexpr int NT1 = N / MT;       // 8192 row tiles

// ---- workspace layout (bytes) ----
constexpr size_t WS_EMB   = 0;                                  // N*DOUT f32
constexpr size_t WS_W0B   = WS_EMB   + (size_t)N * DOUT * 4;    // H*NFEAT bf16
constexpr size_t WS_W1B   = WS_W0B   + (size_t)H * NFEAT * 2;   // DOUT*H bf16
constexpr size_t WS_SCORE = WS_W1B   + (size_t)DOUT * H * 2;    // NTR f32
constexpr size_t WS_GSUM  = WS_SCORE + (size_t)NTR * 4;         // H f32
constexpr size_t WS_GSQ   = WS_GSUM  + (size_t)H * 4;           // H f32
constexpr size_t WS_SCALE = WS_GSQ   + (size_t)H * 4;           // H f32
constexpr size_t WS_SHIFT = WS_SCALE + (size_t)H * 4;           // H f32
constexpr size_t WS_CMAX  = WS_SHIFT + (size_t)H * 4;           // C u32
constexpr size_t WS_DEN   = WS_CMAX  + (size_t)C * 4;           // C f32
constexpr size_t WS_NUM   = WS_DEN   + (size_t)C * 4;           // C*DOUT f32
constexpr size_t WS_PROTO = WS_NUM   + (size_t)C * DOUT * 4;    // C*DOUT f32
constexpr size_t WS_PN    = WS_PROTO + (size_t)C * DOUT * 4;    // C f32
constexpr size_t WS_XB    = (WS_PN + (size_t)C * 4 + 255) & ~(size_t)255; // N*NFEAT bf16

// ---------------- WMMA helpers ----------------
DEVI v8f wmma_bf16(v16bf a, v16bf b, v8f c) {
  return __builtin_amdgcn_wmma_f32_16x16x32_bf16(false, a, false, b, (short)0, c,
                                                 false, false);
}

// A fragment (16x32 bf16): lane holds row m = lane%16; hsel = lane/16 selects the
// contiguous K range {hsel*16..+16} (matching the B fragment order).
DEVI v16bf load_a(const bf16* base, int ld, int m, int kc, int hsel) {
  const v8bf lo = *(const v8bf*)(base + m * ld + kc + hsel * 16);
  const v8bf hi = *(const v8bf*)(base + m * ld + kc + hsel * 16 + 8);
  v16bf a;
#pragma unroll
  for (int e = 0; e < 8; ++e) { a[e] = lo[e]; a[8 + e] = hi[e]; }
  return a;
}

// TDM: DMA one 32-row x 512-byte bf16 tile (dense in global) into LDS, inserting
// 16B of padding after every 512B row (pad_interval=128 dwords -> code 6,
// pad_amount=4 dwords -> code 3) to reproduce the LDA=264 bank-padded layout.
DEVI void tdm_load_tile(const bf16* gsrc, unsigned lds_off) {
  const unsigned long long ga = (unsigned long long)(size_t)gsrc;
  v4u g0;
  g0[0] = 1u;                                        // count=1, user descriptor
  g0[1] = lds_off;                                   // lds_addr
  g0[2] = (unsigned)(ga & 0xFFFFFFFFull);            // global_addr[31:0]
  g0[3] = (unsigned)((ga >> 32) & 0x01FFFFFFull)     // global_addr[56:32]
          | (2u << 30);                              // type=2 (image)
  v8i g1;
  g1[0] = (int)((1u << 20) | (6u << 22) | (3u << 25)); // pad_en, intv=128dw, amt=4dw
  g1[1] = (int)(512u << 16);                         // tensor_dim0 = 512 (bytes)
  g1[2] = (int)(32u << 16);                          // tensor_dim1 = 32 (rows)
  g1[3] = (int)(512u << 16);                         // tile_dim0 = 512
  g1[4] = 32;                                        // tile_dim1 = 32, tile_dim2=0
  g1[5] = 512;                                       // tensor_dim0_stride = 512
  g1[6] = 0;
  g1[7] = 0;
  const v4i z4 = {0, 0, 0, 0};
#if defined(__clang_major__) && (__clang_major__ >= 23)
  const v8i z8 = {0, 0, 0, 0, 0, 0, 0, 0};
  __builtin_amdgcn_tensor_load_to_lds(g0, g1, z4, z4, z8, 0);
#else
  __builtin_amdgcn_tensor_load_to_lds(g0, g1, z4, z4, 0);
#endif
}

// float-order-preserving encode for atomic max on unsigned
DEVI unsigned fenc(float f) {
  unsigned u = __float_as_uint(f);
  return (u & 0x80000000u) ? ~u : (u | 0x80000000u);
}
DEVI float fdec(unsigned u) {
  unsigned b = (u & 0x80000000u) ? (u & 0x7FFFFFFFu) : ~u;
  return __uint_as_float(b);
}

// ---------------- kernels ----------------
__global__ void k_init(float* gsum, float* gsq, float* numer, float* denom,
                       unsigned* cmax) {
  const int t = threadIdx.x;
  for (int j = t; j < H; j += 256) { gsum[j] = 0.f; gsq[j] = 0.f; }
  for (int j = t; j < C * DOUT; j += 256) numer[j] = 0.f;
  for (int j = t; j < C; j += 256) { denom[j] = 0.f; cmax[j] = 0u; }
}

__global__ void k_prep(const float* __restrict__ W0, const float* __restrict__ W1,
                       bf16* __restrict__ W0b, bf16* __restrict__ W1b) {
  const int tot0 = H * NFEAT, tot1 = DOUT * H;
  for (int t = blockIdx.x * 256 + threadIdx.x; t < tot0 + tot1;
       t += gridDim.x * 256) {
    if (t < tot0) W0b[t] = (bf16)W0[t];
    else          W1b[t - tot0] = (bf16)W1[t - tot0];
  }
}

// x (f32) -> xb (bf16), row-major, so GEMM tiles are dense 16KB regions for TDM.
__global__ __launch_bounds__(256) void k_prepx(const float* __restrict__ x,
                                               bf16* __restrict__ xb) {
  const size_t tot = (size_t)N * NFEAT / 4;
  for (size_t t = (size_t)blockIdx.x * 256 + threadIdx.x; t < tot;
       t += (size_t)gridDim.x * 256) {
    const float4 f = ((const float4*)x)[t];
    v4bf o;
    o[0] = (bf16)f.x; o[1] = (bf16)f.y; o[2] = (bf16)f.z; o[3] = (bf16)f.w;
    ((v4bf*)xb)[t] = o;
  }
}

// Pass 1: h = x@W0^T + b0 tiles via WMMA; accumulate per-column sum / sumsq.
__global__ __launch_bounds__(256) void k_stats(const bf16* __restrict__ xb,
                                               const bf16* __restrict__ W0b,
                                               const float* __restrict__ b0,
                                               float* __restrict__ gsum,
                                               float* __restrict__ gsq) {
  __shared__ __align__(16) bf16 xt[MT * LDA];
  __shared__ float lsum[H], lsq[H];
  const int tid = threadIdx.x;
  for (int j = tid; j < H; j += 256) { lsum[j] = 0.f; lsq[j] = 0.f; }
  const int wave = tid >> 5, lane = tid & 31;
  const int hsel = lane >> 4, n = lane & 15;
  for (int tile = blockIdx.x; tile < NT1; tile += gridDim.x) {
    __syncthreads();
    const size_t row0 = (size_t)tile * MT;
    if (tid < 32) {  // wave 0 drives the TDM; EXEC is ignored by tensor ops
      tdm_load_tile(xb + row0 * NFEAT, (unsigned)(size_t)&xt[0]);
      __builtin_amdgcn_s_wait_tensorcnt(0);
    }
    __syncthreads();
    for (int nbl = 0; nbl < 4; ++nbl) {
      const int nb = wave * 4 + nbl;
      const int col = nb * 16 + n;
      const bf16* wr = W0b + (size_t)col * NFEAT;
      v8f acc[2];
#pragma unroll
      for (int mb = 0; mb < 2; ++mb) acc[mb] = (v8f){0, 0, 0, 0, 0, 0, 0, 0};
#pragma unroll
      for (int kc = 0; kc < NFEAT; kc += 32) {
        const v16bf b = *(const v16bf*)(wr + kc + hsel * 16);  // reused by both mb
        acc[0] = wmma_bf16(load_a(xt, LDA, n, kc, hsel), b, acc[0]);
        acc[1] = wmma_bf16(load_a(xt + 16 * LDA, LDA, n, kc, hsel), b, acc[1]);
      }
      const float bv = b0[col];
      float s = 0.f, q = 0.f;
#pragma unroll
      for (int mb = 0; mb < 2; ++mb)
#pragma unroll
        for (int r = 0; r < 8; ++r) {
          const float h = acc[mb][r] + bv;
          s += h; q += h * h;
        }
      s += __shfl_xor(s, 16);
      q += __shfl_xor(q, 16);
      if (hsel == 0) { atomicAdd(&lsum[col], s); atomicAdd(&lsq[col], q); }
    }
  }
  __syncthreads();
  for (int j = tid; j < H; j += 256) {
    atomicAdd(&gsum[j], lsum[j]);
    atomicAdd(&gsq[j], lsq[j]);
  }
}

__global__ void k_bnfin(const float* gsum, const float* gsq, const float* gamma,
                        const float* beta, const float* b0, float* scale,
                        float* shift) {
  const int j = blockIdx.x * blockDim.x + threadIdx.x;
  if (j < H) {
    const float invN = 1.f / (float)N;
    const float mu = gsum[j] * invN;
    const float var = gsq[j] * invN - mu * mu;
    const float sc = gamma[j] * rsqrtf(var + EPS);
    scale[j] = sc;
    shift[j] = beta[j] + (b0[j] - mu) * sc;   // (acc+b0-mu)*sc+beta, b0 folded in
  }
}

// Pass 2: fused GEMM1 -> BN+LeakyReLU (regs) -> bf16 h in LDS -> GEMM2 -> emb f32
__global__ __launch_bounds__(256) void k_fused(const bf16* __restrict__ xb,
                                               const bf16* __restrict__ W0b,
                                               const bf16* __restrict__ W1b,
                                               const float* __restrict__ scale,
                                               const float* __restrict__ shift,
                                               const float* __restrict__ b1,
                                               float* __restrict__ emb) {
  __shared__ __align__(16) bf16 xt[MT * LDA];
  __shared__ __align__(16) bf16 ht[MT * LDH];
  const int tid = threadIdx.x;
  const int wave = tid >> 5, lane = tid & 31;
  const int hsel = lane >> 4, n = lane & 15;
  for (int tile = blockIdx.x; tile < NT1; tile += gridDim.x) {
    __syncthreads();
    const size_t row0 = (size_t)tile * MT;
    if (tid < 32) {
      tdm_load_tile(xb + row0 * NFEAT, (unsigned)(size_t)&xt[0]);
      __builtin_amdgcn_s_wait_tensorcnt(0);
    }
    __syncthreads();
    // GEMM1 + BN + LeakyReLU -> ht (bf16)
    for (int nbl = 0; nbl < 4; ++nbl) {
      const int nb = wave * 4 + nbl;
      const int col = nb * 16 + n;
      const bf16* wr = W0b + (size_t)col * NFEAT;
      v8f acc[2];
#pragma unroll
      for (int mb = 0; mb < 2; ++mb) acc[mb] = (v8f){0, 0, 0, 0, 0, 0, 0, 0};
#pragma unroll
      for (int kc = 0; kc < NFEAT; kc += 32) {
        const v16bf b = *(const v16bf*)(wr + kc + hsel * 16);
        acc[0] = wmma_bf16(load_a(xt, LDA, n, kc, hsel), b, acc[0]);
        acc[1] = wmma_bf16(load_a(xt + 16 * LDA, LDA, n, kc, hsel), b, acc[1]);
      }
      const float sc = scale[col], sh = shift[col];
#pragma unroll
      for (int mb = 0; mb < 2; ++mb)
#pragma unroll
        for (int r = 0; r < 8; ++r) {
          float v = acc[mb][r] * sc + sh;
          v = (v >= 0.f) ? v : SLOPE * v;
          ht[(mb * 16 + r + hsel * 8) * LDH + col] = (bf16)v;
        }
    }
    __syncthreads();
    // GEMM2: emb tile = lrelu(h) @ W1^T + b1
    {
      const int col = wave * 16 + n;
      const bf16* wr = W1b + (size_t)col * H;
      v8f acc[2];
#pragma unroll
      for (int mb = 0; mb < 2; ++mb) acc[mb] = (v8f){0, 0, 0, 0, 0, 0, 0, 0};
#pragma unroll
      for (int kc = 0; kc < H; kc += 32) {
        const v16bf b = *(const v16bf*)(wr + kc + hsel * 16);
        acc[0] = wmma_bf16(load_a(ht, LDH, n, kc, hsel), b, acc[0]);
        acc[1] = wmma_bf16(load_a(ht + 16 * LDH, LDH, n, kc, hsel), b, acc[1]);
      }
      const float bb = b1[col];
#pragma unroll
      for (int mb = 0; mb < 2; ++mb)
#pragma unroll
        for (int r = 0; r < 8; ++r) {
          const size_t row = row0 + mb * 16 + r + hsel * 8;
          emb[row * DOUT + col] = acc[mb][r] + bb;
        }
    }
  }
}

// Attention score for each training row against its OWN class MLP only.
__global__ __launch_bounds__(128) void k_att(const float* __restrict__ emb,
                                             const int* __restrict__ idx_tr,
                                             const int* __restrict__ labels,
                                             const float* __restrict__ aW1,
                                             const float* __restrict__ ab1,
                                             const float* __restrict__ aW2,
                                             const float* __restrict__ ab2,
                                             float* __restrict__ sOut,
                                             unsigned* __restrict__ cmax) {
  __shared__ float er[DOUT];
  __shared__ float red[128];
  const int tid = threadIdx.x;
  for (int row = blockIdx.x; row < NTR; row += gridDim.x) {
    const int i = idx_tr[row];
    const int c = labels[i];
    __syncthreads();
    er[tid] = emb[(size_t)i * DOUT + tid];
    __syncthreads();
    const float* wr = aW1 + ((size_t)c * D + tid) * DOUT;
    float acc = ab1[c * D + tid];
#pragma unroll 8
    for (int k = 0; k < DOUT; ++k) acc += er[k] * wr[k];
    red[tid] = tanhf(acc) * aW2[c * D + tid];
    __syncthreads();
    for (int off = 64; off > 0; off >>= 1) {
      if (tid < off) red[tid] += red[tid + off];
      __syncthreads();
    }
    if (tid == 0) {
      const float sv = red[0] + ab2[c];
      sOut[row] = sv;
      atomicMax(&cmax[c], fenc(sv));
    }
  }
}

// Accumulate per-class sum(exp(s-max)) and sum(exp(s-max)*emb) (LDS, then flush).
__global__ __launch_bounds__(256) void k_pool(const float* __restrict__ emb,
                                              const int* __restrict__ idx_tr,
                                              const int* __restrict__ labels,
                                              const float* __restrict__ s,
                                              const unsigned* __restrict__ cmax,
                                              float* __restrict__ numer,
                                              float* __restrict__ denom) {
  __shared__ float ln[C * DOUT];
  __shared__ float ld[C];
  const int tid = threadIdx.x, wave = tid >> 5, lane = tid & 31;
  for (int j = tid; j < C * DOUT; j += 256) ln[j] = 0.f;
  for (int j = tid; j < C; j += 256) ld[j] = 0.f;
  __syncthreads();
  for (int row = blockIdx.x * 8 + wave; row < NTR; row += gridDim.x * 8) {
    const int i = idx_tr[row];
    const int c = labels[i];
    const float w = __expf(s[row] - fdec(cmax[c]));
    const float* e = emb + (size_t)i * DOUT;
#pragma unroll
    for (int j = 0; j < 4; ++j) {
      const int d = lane + j * 32;
      atomicAdd(&ln[c * DOUT + d], w * e[d]);
    }
    if (lane == 0) atomicAdd(&ld[c], w);
  }
  __syncthreads();
  for (int j = tid; j < C * DOUT; j += 256) atomicAdd(&numer[j], ln[j]);
  for (int j = tid; j < C; j += 256) atomicAdd(&denom[j], ld[j]);
}

// protos = numer/denom, ||p||^2, and proto_dist = sum(sqdist(p,p))/(C*(C-1)/2)
__global__ __launch_bounds__(256) void k_proto(const float* __restrict__ numer,
                                               const float* __restrict__ denom,
                                               float* __restrict__ proto,
                                               float* __restrict__ pn,
                                               float* __restrict__ out_pd) {
  __shared__ float sp[C * DOUT];
  __shared__ float spn[C];
  __shared__ float red[256];
  const int tid = threadIdx.x;
  for (int j = tid; j < C * DOUT; j += 256) {
    const float v = numer[j] / denom[j >> 7];
    sp[j] = v;
    proto[j] = v;
  }
  __syncthreads();
  for (int c = tid; c < C; c += 256) {
    float s = 0.f;
    for (int k = 0; k < DOUT; ++k) s += sp[c * DOUT + k] * sp[c * DOUT + k];
    spn[c] = s;
    pn[c] = s;
  }
  __syncthreads();
  float local = 0.f;
  for (int p = tid; p < C * C; p += 256) {
    const int a = p >> 5, b = p & 31;
    float dot = 0.f;
    for (int k = 0; k < DOUT; ++k) dot += sp[a * DOUT + k] * sp[b * DOUT + k];
    local += spn[a] + spn[b] - 2.f * dot;
  }
  red[tid] = local;
  __syncthreads();
  for (int off = 128; off > 0; off >>= 1) {
    if (tid < off) red[tid] += red[tid + off];
    __syncthreads();
  }
  if (tid == 0) *out_pd = red[0] / (float)(C * (C - 1) / 2);
}

// -dists = -(||e||^2 + ||p||^2 - 2 e.p) for all (row, class)
__global__ __launch_bounds__(256) void k_dists(const float* __restrict__ emb,
                                               const float* __restrict__ proto,
                                               const float* __restrict__ pn,
                                               float* __restrict__ out) {
  __shared__ float sp[C * DOUT];
  __shared__ float spn[C];
  const int tid = threadIdx.x;
  for (int j = tid; j < C * DOUT; j += 256) sp[j] = proto[j];
  for (int j = tid; j < C; j += 256) spn[j] = pn[j];
  __syncthreads();
  const size_t total = (size_t)N * C;
  for (size_t t = (size_t)blockIdx.x * 256 + tid; t < total;
       t += (size_t)gridDim.x * 256) {
    const size_t row = t >> 5;
    const int c = (int)(t & 31);
    const float* e = emb + row * DOUT;
    const float* p = sp + c * DOUT;
    float dot = 0.f, en = 0.f;
#pragma unroll 8
    for (int k = 0; k < DOUT; ++k) {
      const float v = e[k];
      dot += v * p[k];
      en += v * v;
    }
    out[t] = -(en + spn[c] - 2.f * dot);
  }
}

extern "C" void kernel_launch(void* const* d_in, const int* in_sizes, int n_in,
                              void* d_out, int out_size, void* d_ws, size_t ws_size,
                              hipStream_t stream) {
  (void)in_sizes; (void)n_in; (void)out_size; (void)ws_size;
  const float* x      = (const float*)d_in[0];
  const int*   labels = (const int*)d_in[1];
  const int*   idx_tr = (const int*)d_in[2];
  const float* W0     = (const float*)d_in[5];
  const float* b0     = (const float*)d_in[6];
  const float* gamma0 = (const float*)d_in[7];
  const float* beta0  = (const float*)d_in[8];
  const float* W1     = (const float*)d_in[9];
  const float* b1     = (const float*)d_in[10];
  const float* aW1    = (const float*)d_in[11];
  const float* ab1    = (const float*)d_in[12];
  const float* aW2    = (const float*)d_in[13];
  const float* ab2    = (const float*)d_in[14];
  float* out = (float*)d_out;

  char* ws = (char*)d_ws;
  float*    emb   = (float*)(ws + WS_EMB);
  bf16*     W0b   = (bf16*)(ws + WS_W0B);
  bf16*     W1b   = (bf16*)(ws + WS_W1B);
  float*    sbuf  = (float*)(ws + WS_SCORE);
  float*    gsum  = (float*)(ws + WS_GSUM);
  float*    gsq   = (float*)(ws + WS_GSQ);
  float*    scale = (float*)(ws + WS_SCALE);
  float*    shift = (float*)(ws + WS_SHIFT);
  unsigned* cmax  = (unsigned*)(ws + WS_CMAX);
  float*    denom = (float*)(ws + WS_DEN);
  float*    numer = (float*)(ws + WS_NUM);
  float*    proto = (float*)(ws + WS_PROTO);
  float*    pn    = (float*)(ws + WS_PN);
  bf16*     xb    = (bf16*)(ws + WS_XB);

  k_init <<<1,    256, 0, stream>>>(gsum, gsq, numer, denom, cmax);
  k_prep <<<256,  256, 0, stream>>>(W0, W1, W0b, W1b);
  k_prepx<<<8192, 256, 0, stream>>>(x, xb);
  k_stats<<<2048, 256, 0, stream>>>(xb, W0b, b0, gsum, gsq);
  k_bnfin<<<2,    256, 0, stream>>>(gsum, gsq, gamma0, beta0, b0, scale, shift);
  k_fused<<<4096, 256, 0, stream>>>(xb, W0b, W1b, scale, shift, b1, emb);
  k_att  <<<4096, 128, 0, stream>>>(emb, idx_tr, labels, aW1, ab1, aW2, ab2, sbuf, cmax);
  k_pool <<<256,  256, 0, stream>>>(emb, idx_tr, labels, sbuf, cmax, numer, denom);
  k_proto<<<1,    256, 0, stream>>>(numer, denom, proto, pn, out + (size_t)N * C);
  k_dists<<<4096, 256, 0, stream>>>(emb, proto, pn, out);
}
